// RNN_LM_54417235640708
// MI455X (gfx1250) — compile-verified, matrix operand order checked by reading, not run on hardware
//
#include <hip/hip_runtime.h>
#include <hip/hip_bf16.h>

#define PAD_IDX 0

#ifndef __has_builtin
#define __has_builtin(x) 0
#endif

// Tensor Data Mover available? (device pass only; host pass parses fallback)
#if defined(__AMDGCN__) && __has_builtin(__builtin_amdgcn_tensor_load_to_lds) && \
    __has_builtin(__builtin_amdgcn_s_wait_tensorcnt)
#define USE_TDM 1
#else
#define USE_TDM 0
#endif

namespace {
constexpr int  kE = 1024, kH = 1024, kL = 2, kO = 32000;
constexpr int  kB = 16, kS = 512;
constexpr int  kM = kB * kS;                       // 8192 activation rows
constexpr long kLogits = (long)kB * kS * kO;       // floats in logits output
}

typedef __attribute__((ext_vector_type(16))) __bf16 v16bf;
typedef __attribute__((ext_vector_type(8)))  float  v8f;
typedef __attribute__((ext_vector_type(4))) unsigned int u32x4;
typedef __attribute__((ext_vector_type(4))) int          i32x4;
typedef __attribute__((ext_vector_type(8))) int          i32x8;

struct U32x8 { uint4 lo, hi; };

// 16 consecutive bf16 (32 bytes, 16B-aligned) -> WMMA operand fragment
__device__ __forceinline__ v16bf load_frag16(const unsigned short* p) {
  U32x8 u;
  u.lo = *reinterpret_cast<const uint4*>(p);
  u.hi = *reinterpret_cast<const uint4*>(p + 8);
  return __builtin_bit_cast(v16bf, u);
}

__device__ __forceinline__ unsigned short f2bf(float x) {
  return __builtin_bit_cast(unsigned short, __float2bfloat16(x));
}

#if USE_TDM
// ---- Tensor DMA descriptor construction (CDNA5 ISA ch.8, D# groups) ----
__device__ __forceinline__ unsigned lds_lo32(const void* p) {
  // LDS aperture: flat addr[31:0] is the DS byte offset
  return (unsigned)(unsigned long long)(uintptr_t)p;
}

__device__ __forceinline__ u32x4 tdm_group0(unsigned ldsAddr, unsigned long long gaddr) {
  u32x4 g;
  g.x = 1u;                                                     // count=1, no gather
  g.y = ldsAddr;                                                // lds_addr (bytes)
  g.z = (unsigned)gaddr;                                        // global_addr[31:0]
  g.w = ((unsigned)(gaddr >> 32) & 0x01FFFFFFu) | (2u << 30);   // addr[56:32] | type=2
  return g;
}

// 2D bf16 tile: tileD0 x tileD1 elements, tensor row stride = stride0 elements.
// LDS padding: +4 DWORDs after every 16 DWORDs => 64B data row + 16B pad = 80B
// LDS row, i.e. exactly the LDP=40-halves bank-conflict-free layout.
__device__ __forceinline__ i32x8 tdm_group1(unsigned tD0, unsigned tD1,
                                            unsigned tileD0, unsigned tileD1,
                                            unsigned long long stride0) {
  i32x8 g;
  g[0] = (1 << 16)            // data_size = 2 bytes
       | (1 << 20)            // pad_enable
       | (3 << 22)            // pad_interval: every 16 DWORDs
       | (3 << 25);           // pad_amount: 4 DWORDs
  g[1] = (int)((tD0 & 0xFFFFu) << 16);                               // [63:48]=tensor_dim0 lo
  g[2] = (int)(((tD0 >> 16) & 0xFFFFu) | ((tD1 & 0xFFFFu) << 16));   // dim0 hi | dim1 lo
  g[3] = (int)(((tD1 >> 16) & 0xFFFFu) | ((tileD0 & 0xFFFFu) << 16));// dim1 hi | tile_dim0
  g[4] = (int)(tileD1 & 0xFFFFu);                                    // tile_dim1, tile_dim2=0
  g[5] = (int)(stride0 & 0xFFFFFFFFull);                             // tensor_dim0_stride lo
  g[6] = (int)((stride0 >> 32) & 0xFFFFull);                         // stride hi | dim1_stride=0
  g[7] = 0;
  return g;
}

#if __has_include(<hip/amd_detail/amd_gfx1250_TDM.h>)
// therock/clang-23 toolchain: 6-arg builtin
__device__ __forceinline__ void tdm_load(u32x4 g0, i32x8 g1) {
  i32x4 z4 = {0, 0, 0, 0};
  i32x8 z8 = {0, 0, 0, 0, 0, 0, 0, 0};
  __builtin_amdgcn_tensor_load_to_lds(g0, g1, z4, z4, z8, 0);
}
#else
// ROCm 7.2 / clang-22 toolchain: 5-arg builtin
__device__ __forceinline__ void tdm_load(u32x4 g0, i32x8 g1) {
  i32x4 z4 = {0, 0, 0, 0};
  __builtin_amdgcn_tensor_load_to_lds(g0, g1, z4, z4, 0);
}
#endif
#endif  // USE_TDM

// ---------------------------------------------------------------------------
// fp32 -> bf16 weight cast (one-time; weights then live in L2 as bf16)
// ---------------------------------------------------------------------------
__global__ void cast_f32_to_bf16(const float* __restrict__ src,
                                 unsigned short* __restrict__ dst, long n) {
  long i = (long)blockIdx.x * blockDim.x + threadIdx.x;
  const long stride = (long)gridDim.x * blockDim.x;
  for (; i < n; i += stride) dst[i] = f2bf(src[i]);
}

// ---------------------------------------------------------------------------
// Embedding lookup with padding_idx=0 semantics, cast to bf16 [B*S, E]
// ---------------------------------------------------------------------------
__global__ void embed_cast_bf16(const int* __restrict__ tok,
                                const float* __restrict__ emb,
                                unsigned short* __restrict__ xb) {
  const int row = blockIdx.x;                      // 0..B*S-1
  const int t = tok[row];
  const bool pad = (t == PAD_IDX);
  const float* src = emb + (long)t * kE;
  unsigned short* dst = xb + (long)row * kE;
  for (int i = threadIdx.x; i < kE; i += blockDim.x)
    dst[i] = f2bf(pad ? 0.f : src[i]);
}

// ---------------------------------------------------------------------------
// D[m,n] = sum_k A[m,k] * Wt[n,k] + bias0[n] (+ bias1[n])
// A: [M,K] bf16 row-major, Wt: [N,K] bf16 row-major (B operand = Wt^T).
// Block tile 128x128, 8 waves, each wave 4x2 WMMA tiles.
// TDM path: double-buffered LDS, wave0 issues tensor_load_to_lds for step
// kk+1 (A+B tiles, hardware-padded rows), s_wait_tensorcnt 2, barrier, WMMA.
// NTSTORE=true streams the output past L2 (logits) to keep weights resident.
// ---------------------------------------------------------------------------
template <bool NTSTORE>
__global__ void __launch_bounds__(256)
gemm_bf16_wmma(const unsigned short* __restrict__ A,
               const unsigned short* __restrict__ Wt,
               const float* __restrict__ bias0,
               const float* __restrict__ bias1,
               float* __restrict__ D,
               int N, int K) {
  constexpr int LDP = 40;   // padded k-stride (halves): 32 data + 8 pad
  const int tid  = threadIdx.x;
  const int lane = tid & 31;
  const int wave = tid >> 5;
  const int lm   = lane & 15;
  const int kl   = (lane < 16) ? 0 : 16;   // which 16-wide K half this lane holds
  const int mOff = (lane < 16) ? 0 : 8;    // C/D row offset per documented layout
  const int rG   = wave & 1;               // 64-row group within block tile
  const int cG   = wave >> 1;              // 32-col group within block tile

  const long mBlock = (long)blockIdx.y * 128;
  const long nBlock = (long)blockIdx.x * 128;

  v8f acc[4][2] = {};

#if USE_TDM
  __shared__ __align__(16) unsigned short aLds[2][128 * LDP];
  __shared__ __align__(16) unsigned short bLds[2][128 * LDP];

  const int nk = K / 32;
  const unsigned long long aBase =
      (unsigned long long)(uintptr_t)(A + mBlock * (long)K);
  const unsigned long long bBase =
      (unsigned long long)(uintptr_t)(Wt + nBlock * (long)K);
  const i32x8 g1 = tdm_group1((unsigned)K, 128u, 32u, 128u, (unsigned long long)K);

  if (wave == 0) {                                   // prologue: stage k-step 0
    tdm_load(tdm_group0(lds_lo32(&aLds[0][0]), aBase), g1);
    tdm_load(tdm_group0(lds_lo32(&bLds[0][0]), bBase), g1);
  }

  for (int kk = 0; kk < nk; ++kk) {
    const int p = kk & 1;
    if (wave == 0) {
      if (kk + 1 < nk) {                             // prefetch next k-slice (64B step)
        const unsigned long long off = (unsigned long long)(kk + 1) * 64ull;
        tdm_load(tdm_group0(lds_lo32(&aLds[p ^ 1][0]), aBase + off), g1);
        tdm_load(tdm_group0(lds_lo32(&bLds[p ^ 1][0]), bBase + off), g1);
        __builtin_amdgcn_s_wait_tensorcnt(2);        // in-order: buffer p complete
      } else {
        __builtin_amdgcn_s_wait_tensorcnt(0);
      }
    }
    __syncthreads();                                 // publish TDM data to all waves

    v16bf bfrag[2];
#pragma unroll
    for (int nt = 0; nt < 2; ++nt)
      bfrag[nt] = load_frag16(&bLds[p][(cG * 32 + nt * 16 + lm) * LDP + kl]);
#pragma unroll
    for (int mt = 0; mt < 4; ++mt) {
      v16bf afrag = load_frag16(&aLds[p][(rG * 64 + mt * 16 + lm) * LDP + kl]);
#pragma unroll
      for (int nt = 0; nt < 2; ++nt)
        acc[mt][nt] = __builtin_amdgcn_wmma_f32_16x16x32_bf16(
            false, afrag, false, bfrag[nt], (short)0, acc[mt][nt], false, false);
    }
    __syncthreads();            // retire reads of buf p before TDM overwrites it
  }
#else
  // -------- fallback: manual single-buffered staging --------
  __shared__ __align__(16) unsigned short aLds[128 * LDP];
  __shared__ __align__(16) unsigned short bLds[128 * LDP];

  const int sRow  = tid >> 1;              // 0..127
  const int sHalf = (tid & 1) * 16;        // 0 or 16 (halves)
  const unsigned short* aSrc = A  + (mBlock + sRow) * (long)K + sHalf;
  const unsigned short* bSrc = Wt + (nBlock + sRow) * (long)K + sHalf;
  unsigned short* aDst = &aLds[sRow * LDP + sHalf];
  unsigned short* bDst = &bLds[sRow * LDP + sHalf];

  for (int k = 0; k < K; k += 32) {
    *reinterpret_cast<uint4*>(aDst)     = *reinterpret_cast<const uint4*>(aSrc + k);
    *reinterpret_cast<uint4*>(aDst + 8) = *reinterpret_cast<const uint4*>(aSrc + k + 8);
    *reinterpret_cast<uint4*>(bDst)     = *reinterpret_cast<const uint4*>(bSrc + k);
    *reinterpret_cast<uint4*>(bDst + 8) = *reinterpret_cast<const uint4*>(bSrc + k + 8);
    __syncthreads();

    v16bf bfrag[2];
#pragma unroll
    for (int nt = 0; nt < 2; ++nt)
      bfrag[nt] = load_frag16(&bLds[(cG * 32 + nt * 16 + lm) * LDP + kl]);
#pragma unroll
    for (int mt = 0; mt < 4; ++mt) {
      v16bf afrag = load_frag16(&aLds[(rG * 64 + mt * 16 + lm) * LDP + kl]);
#pragma unroll
      for (int nt = 0; nt < 2; ++nt)
        acc[mt][nt] = __builtin_amdgcn_wmma_f32_16x16x32_bf16(
            false, afrag, false, bfrag[nt], (short)0, acc[mt][nt], false, false);
    }
    __syncthreads();
  }
#endif

#pragma unroll
  for (int nt = 0; nt < 2; ++nt) {
    const long n = nBlock + cG * 32 + nt * 16 + lm;
    float bn = bias0[n];
    if (bias1) bn += bias1[n];
#pragma unroll
    for (int mt = 0; mt < 4; ++mt) {
#pragma unroll
      for (int j = 0; j < 8; ++j) {
        const long m = mBlock + rG * 64 + mt * 16 + mOff + j;
        const float v = acc[mt][nt][j] + bn;
        if constexpr (NTSTORE)
          __builtin_nontemporal_store(v, &D[m * (long)N + n]);
        else
          D[m * (long)N + n] = v;
      }
    }
  }
}

// ---------------------------------------------------------------------------
// Elman recurrence for one layer, persistent single workgroup (32 waves).
//   h_t = tanh(pre[:,t,:] + h_{t-1} @ Whh^T)     (biases folded into pre)
// h state double-buffered in LDS (bf16, padded stride); Whh L2-resident.
// Each wave owns 32 columns (2 WMMA n-tiles); one barrier per timestep.
// kdim is a RUNTIME argument on purpose: it stops clang from fully unrolling
// the k-loop, which previously made all W_hh fragment loads t-invariant,
// hoisted ~512 VGPRs of fragments above the time loop and spilled to scratch.
// Runtime bound => loads depend on the induction var => no hoist, no spill.
// ---------------------------------------------------------------------------
__global__ void __launch_bounds__(1024)
rnn_scan_wmma(const float* __restrict__ pre,            // [B*S, H] fp32
              const unsigned short* __restrict__ Whh,   // [H, H] bf16 row-major
              unsigned short* __restrict__ y,           // [B*S, H] bf16 out
              float* __restrict__ hOut,                 // [B, H] final hidden
              int kdim) {                               // == kH (runtime)
  constexpr int HP = kH + 8;                 // padded row stride (halves)
  __shared__ __align__(16) unsigned short hbuf[2][16 * HP];

  const int lane = threadIdx.x & 31;
  const int wave = threadIdx.x >> 5;         // 0..31 -> cols [wave*32, wave*32+32)
  const int lm   = lane & 15;
  const int kl   = (lane < 16) ? 0 : 16;
  const int mOff = (lane < 16) ? 0 : 8;

  for (int i = threadIdx.x; i < 16 * HP; i += blockDim.x) hbuf[0][i] = 0;  // h0 = 0
  __syncthreads();

  int cur = 0;
  for (int t = 0; t < kS; ++t) {
#pragma unroll
    for (int tile = 0; tile < 2; ++tile) {
      const int n = wave * 32 + tile * 16 + lm;
      const unsigned short* bRow = Whh + (long)n * kH + kl;
      const unsigned short* aRow = &hbuf[cur][lm * HP + kl];
      v8f acc = {};
      // outer loop: runtime trip count (no full unroll); inner: 4x unrolled
      // -> clause-batched ds_load/global_load pairs feeding 4 WMMAs.
      for (int k0 = 0; k0 < kdim; k0 += 128) {
#pragma unroll
        for (int ku = 0; ku < 128; ku += 32) {
          const int k = k0 + ku;
          v16bf a = load_frag16(aRow + k);   // LDS (ds_load_b128)
          v16bf b = load_frag16(bRow + k);   // global, L2-resident
          acc = __builtin_amdgcn_wmma_f32_16x16x32_bf16(
              false, a, false, b, (short)0, acc, false, false);
        }
      }
#pragma unroll
      for (int j = 0; j < 8; ++j) {
        const int m = mOff + j;              // batch row 0..15
        const float hv = tanhf(pre[((long)m * kS + t) * kH + n] + acc[j]);
        const unsigned short hb = f2bf(hv);
        hbuf[cur ^ 1][m * HP + n] = hb;
        y[((long)m * kS + t) * kH + n] = hb;
        if (t == kS - 1) hOut[m * kH + n] = hv;
      }
    }
    __syncthreads();                         // publish h_t, retire reads of h_{t-1}
    cur ^= 1;
  }
}

// ---------------------------------------------------------------------------
extern "C" void kernel_launch(void* const* d_in, const int* in_sizes, int n_in,
                              void* d_out, int out_size, void* d_ws, size_t ws_size,
                              hipStream_t stream) {
  const int*   tok   = (const int*)d_in[0];
  const float* emb   = (const float*)d_in[1];
  const float* w_ih  = (const float*)d_in[2];   // [L,H,E]
  const float* w_hh  = (const float*)d_in[3];   // [L,H,H]
  const float* b_ih  = (const float*)d_in[4];   // [L,H]
  const float* b_hh  = (const float*)d_in[5];   // [L,H]
  const float* w_out = (const float*)d_in[6];   // [O,H]
  const float* b_out = (const float*)d_in[7];   // [O]
  float* out = (float*)d_out;                   // logits [B,S,O] then h_n [L,B,H]

  // Workspace carve-up (~135 MB total)
  char* ws = (char*)d_ws;
  unsigned short* xb    = (unsigned short*)ws;                    // [M,E] bf16  16.8 MB
  unsigned short* yb    = xb + (long)kM * kE;                     // [M,H] bf16  16.8 MB
  float*          pre   = (float*)(yb + (long)kM * kH);           // [M,H] f32   33.6 MB
  unsigned short* wihb  = (unsigned short*)(pre + (long)kM * kH); // [L,H,E]      4.2 MB
  unsigned short* whhb  = wihb + (long)kL * kH * kE;              // [L,H,H]      4.2 MB
  unsigned short* woutb = whhb + (long)kL * kH * kH;              // [O,H]       65.5 MB

  // 1) one-time bf16 weight casts (stay hot in 192 MB L2)
  cast_f32_to_bf16<<<2048, 256, 0, stream>>>(w_ih,  wihb,  (long)kL * kH * kE);
  cast_f32_to_bf16<<<2048, 256, 0, stream>>>(w_hh,  whhb,  (long)kL * kH * kH);
  cast_f32_to_bf16<<<4096, 256, 0, stream>>>(w_out, woutb, (long)kO * kH);

  // 2) embedding -> bf16 activations
  embed_cast_bf16<<<kM, 256, 0, stream>>>(tok, emb, xb);

  // 3) layer 0: input projection (b_ih + b_hh folded), then serial scan
  gemm_bf16_wmma<false><<<dim3(kH / 128, kM / 128), 256, 0, stream>>>(
      xb, wihb, b_ih, b_hh, pre, kH, kE);
  rnn_scan_wmma<<<1, 1024, 0, stream>>>(pre, whhb, yb, out + kLogits, kH);

  // 4) layer 1
  gemm_bf16_wmma<false><<<dim3(kH / 128, kM / 128), 256, 0, stream>>>(
      yb, wihb + (long)kH * kE, b_ih + kH, b_hh + kH, pre, kH, kH);
  rnn_scan_wmma<<<1, 1024, 0, stream>>>(
      pre, whhb + (long)kH * kH, yb, out + kLogits + (long)kB * kH, kH);

  // 5) output projection: the 537 GFLOP GEMM -> fp32 logits, streamed past L2
  gemm_bf16_wmma<true><<<dim3(kO / 128, kM / 128), 256, 0, stream>>>(
      yb, woutb, b_out, nullptr, out, kO, kH);
}
